// SpatialCrossAttention_69209103008007
// MI455X (gfx1250) — matrix operand order repaired and checked by reading, not compiled, not run on hardware
//
#include <hip/hip_runtime.h>
#include <hip/hip_bf16.h>

typedef __attribute__((ext_vector_type(16))) __bf16 v16bf;
typedef __attribute__((ext_vector_type(8)))  float  v8f;
typedef __attribute__((ext_vector_type(8)))  __bf16 bf16x8;

#define NQC   2500
#define DIMC  256
#define NCAM  6
#define NHEAD 8
#define NPT   4
#define HFI   64
#define WFI   176
#define HWC   (HFI*WFI)           // 11264 (multiple of 128)
#define NOFF  (NCAM*NHEAD*NPT*2)  // 384
#define NLG   (NCAM*NHEAD*NPT)    // 192
#define NQO   (NOFF+NLG)          // 576

// ---------------- elementwise fp32 -> bf16 (row-major A operands) ----------------
__global__ void cvt_f32_bf16(const float* __restrict__ s, __bf16* __restrict__ d, int n) {
    int i = blockIdx.x * 256 + threadIdx.x;
    if (i < n) d[i] = (__bf16)s[i];
}

// W[k][n] fp32 -> Wt[n][k] bf16 (k-contiguous for LDS staging)
__global__ void cvt_transpose_w(const float* __restrict__ W, __bf16* __restrict__ Wt,
                                int Kd, int Nd) {
    int i = blockIdx.x * 256 + threadIdx.x;
    if (i < Kd * Nd) {
        int n = i / Kd, k = i % Kd;
        Wt[i] = (__bf16)W[k * Nd + n];
    }
}

// Pack W_off|W_w -> transposed [576][256] bf16 + fused bias
__global__ void pack_qweights_t(const float* __restrict__ Woff, const float* __restrict__ Ww,
                                const float* __restrict__ boff, const float* __restrict__ bw,
                                __bf16* __restrict__ Wqt, float* __restrict__ biasq) {
    int i = blockIdx.x * 256 + threadIdx.x;
    if (i < NQO * DIMC) {
        int n = i / DIMC, k = i % DIMC;
        float v = (n < NOFF) ? Woff[k * NOFF + n] : Ww[k * NLG + (n - NOFF)];
        Wqt[i] = (__bf16)v;
    }
    if (i < NQO) biasq[i] = (i < NOFF) ? boff[i] : bw[i - NOFF];
}

// img [cam][256ch][HW] fp32 -> imgT [cam][HW][256ch] bf16 (32x32 LDS tile transpose)
__global__ __launch_bounds__(256)
void transpose_cvt_img(const float* __restrict__ img, __bf16* __restrict__ imgT) {
    __shared__ __bf16 tile[32][33];
    const int c   = blockIdx.z;
    const int hw0 = blockIdx.x * 32;
    const int ch0 = blockIdx.y * 32;
    const int tid = threadIdx.x;
    #pragma unroll
    for (int it = 0; it < 4; ++it) {
        int i = tid + it * 256;
        int ch = i >> 5, hw = i & 31;    // consecutive tid -> consecutive hw (coalesced)
        tile[ch][hw] = (__bf16)img[((size_t)c * DIMC + ch0 + ch) * HWC + hw0 + hw];
    }
    __syncthreads();
    #pragma unroll
    for (int it = 0; it < 4; ++it) {
        int i = tid + it * 256;
        int hw = i >> 5, ch = i & 31;    // consecutive tid -> consecutive ch (coalesced)
        imgT[((size_t)c * HWC + hw0 + hw) * DIMC + ch0 + ch] = tile[ch][hw];
    }
}

// ---------------- bf16 WMMA GEMM (NT): C[m,n] = sum_k A[m,k]*Bt[n,k] + bias[n] ----
// A: bf16 [M][256] row-major. Bt: bf16 [N][256]. Both k-contiguous -> b128 staging,
// and fragments read as 2x ds_load_b128 per matrix (LDS rows padded to 40 bf16 =
// 20 banks -> conflict-free). 8 waves x 4 v_wmma_f32_16x16x32_bf16 per K-step.
template <bool OUT_BF16>
__global__ __launch_bounds__(256)
void gemm_nt_wmma(const __bf16* __restrict__ A, const __bf16* __restrict__ Bt,
                  const float* __restrict__ bias, void* __restrict__ Cv,
                  int M, int N, long long strideA, long long strideC) {
    constexpr int K   = 256;
    constexpr int LDS = 40;                  // 32 + 8 pad
    __shared__ __bf16 a_s[128 * LDS];        // [m][k]
    __shared__ __bf16 b_s[64 * LDS];         // [n][k]

    const int tid  = threadIdx.x;
    const int lane = tid & 31;
    const int wave = tid >> 5;
    const int half = lane >> 4;
    const int lm   = lane & 15;
    const int m0   = blockIdx.x * 128;
    const int n0   = blockIdx.y * 64;
    const __bf16* Ap = A + (long long)blockIdx.z * strideA;

    v8f acc[4] = {};

    for (int k0 = 0; k0 < K; k0 += 32) {
        // stage A tile 128x32 (2 x bf16x8 per thread, coalesced)
        #pragma unroll
        for (int it = 0; it < 2; ++it) {
            int i  = tid + it * 256;
            int mm = i >> 2, kb = (i & 3) * 8;
            int gm = m0 + mm;
            bf16x8 v;
            if (gm < M) {
                v = *(const bf16x8*)&Ap[(size_t)gm * K + k0 + kb];
            } else {
                #pragma unroll
                for (int t = 0; t < 8; ++t) v[t] = (__bf16)0.f;
            }
            *(bf16x8*)&a_s[mm * LDS + kb] = v;
        }
        // stage B tile 64x32 (1 x bf16x8 per thread)
        {
            int nn = tid >> 2, kb = (tid & 3) * 8;
            *(bf16x8*)&b_s[nn * LDS + kb] =
                *(const bf16x8*)&Bt[(size_t)(n0 + nn) * K + k0 + kb];
        }
        __syncthreads();

        // A fragment: k = {half*8..+7} U {16+half*8..+7} at fixed m -> 2 b128 loads
        union { v16bf v; bf16x8 h[2]; } af;
        const int arow = (wave << 4) + lm;
        af.h[0] = *(const bf16x8*)&a_s[arow * LDS + (half << 3)];
        af.h[1] = *(const bf16x8*)&a_s[arow * LDS + 16 + (half << 3)];

        #pragma unroll
        for (int nt = 0; nt < 4; ++nt) {
            // B fragment: k = {half*8..+7} U {16+half*8..+7} at fixed n -> 2 b128 loads
            union { v16bf v; bf16x8 h[2]; } bfr;
            const int brow = (nt << 4) + lm;
            bfr.h[0] = *(const bf16x8*)&b_s[brow * LDS + (half << 3)];
            bfr.h[1] = *(const bf16x8*)&b_s[brow * LDS + 16 + (half << 3)];
            acc[nt] = __builtin_amdgcn_wmma_f32_16x16x32_bf16(
                false, af.v, false, bfr.v, (short)0, acc[nt], false, false);
        }
        __syncthreads();
    }

    // D layout: VGPR r -> M = r + 8*half, N = lane&15
    #pragma unroll
    for (int nt = 0; nt < 4; ++nt) {
        #pragma unroll
        for (int r = 0; r < 8; ++r) {
            int m = m0 + (wave << 4) + (half << 3) + r;
            int n = n0 + (nt << 4) + lm;
            if (m < M) {
                float v = acc[nt][r] + bias[n];
                if (OUT_BF16) ((__bf16*)Cv + (long long)blockIdx.z * strideC)[(size_t)m * N + n] = (__bf16)v;
                else          ((float*)Cv  + (long long)blockIdx.z * strideC)[(size_t)m * N + n] = v;
            }
        }
    }
}

// ---------------- softmax + bilinear gather + weighted accumulate ----------------
__device__ __forceinline__ float wred_max(float v) {
    #pragma unroll
    for (int o = 16; o > 0; o >>= 1) v = fmaxf(v, __shfl_xor(v, o, 32));
    return v;
}
__device__ __forceinline__ float wred_sum(float v) {
    #pragma unroll
    for (int o = 16; o > 0; o >>= 1) v += __shfl_xor(v, o, 32);
    return v;
}

__global__ __launch_bounds__(256)
void sample_accum(const float* __restrict__ qout, const float* __restrict__ ref,
                  const unsigned char* __restrict__ vmask, const __bf16* __restrict__ valb,
                  __bf16* __restrict__ outpre) {
    const int q   = blockIdx.x;
    const int tid = threadIdx.x;
    __shared__ float s_logit[NLG];
    __shared__ float s_red[8];
    __shared__ float s_cw[NLG * 4];
    __shared__ int   s_cl[NLG * 4];

    const float* qrow = qout + (size_t)q * NQO;
    const float NINF = -__builtin_inff();

    if (tid < NLG) {
        int c = tid >> 5;                    // 32 = NH*NP points per camera
        float lg = qrow[NOFF + tid];
        s_logit[tid] = (vmask[c * NQC + q] != 0) ? lg : NINF;
    }
    __syncthreads();

    float lv = (tid < NLG) ? s_logit[tid] : NINF;
    float wm = wred_max(lv);
    if ((tid & 31) == 0) s_red[tid >> 5] = wm;
    __syncthreads();
    float maxv = NINF;
    #pragma unroll
    for (int i = 0; i < 8; ++i) maxv = fmaxf(maxv, s_red[i]);

    float ev = (tid < NLG && maxv != NINF) ? __expf(lv - maxv) : 0.f;
    float psum = wred_sum(ev);
    __syncthreads();
    if ((tid & 31) == 0) s_red[tid >> 5] = psum;
    __syncthreads();
    float sum = 0.f;
    #pragma unroll
    for (int i = 0; i < 8; ++i) sum += s_red[i];
    float attw = (sum > 0.f) ? (ev / sum) : 0.f;   // nan_to_num(all -inf) -> 0

    if (tid < NLG) {
        int c = tid >> 5;
        float offx = qrow[2 * tid + 0];
        float offy = qrow[2 * tid + 1];
        float px = ref[((size_t)c * NQC + q) * 2 + 0] + offx * (2.f / (WFI - 1));
        float py = ref[((size_t)c * NQC + q) * 2 + 1] + offy * (2.f / (HFI - 1));
        float x = (px + 1.f) * 0.5f * (WFI - 1);
        float y = (py + 1.f) * 0.5f * (HFI - 1);
        float x0f = floorf(x), y0f = floorf(y);
        float wx = x - x0f, wy = y - y0f;
        int x0 = (int)x0f, y0 = (int)y0f;
        #pragma unroll
        for (int j = 0; j < 4; ++j) {
            int xi = x0 + (j & 1);
            int yi = y0 + (j >> 1);
            bool v = (xi >= 0) & (xi < WFI) & (yi >= 0) & (yi < HFI);
            float cw = v ? attw * ((j & 1) ? wx : 1.f - wx) * ((j >> 1) ? wy : 1.f - wy) : 0.f;
            s_cw[tid * 4 + j] = cw;
            s_cl[tid * 4 + j] = v ? (c * HWC + yi * WFI + xi) : 0;
        }
    }
    __syncthreads();

    // one channel per thread; each corner is a coalesced 512B row in valb (L2-resident)
    float acc = 0.f;
    const int e = tid;
    for (int i = 0; i < NLG * 4; ++i) {
        float cw = s_cw[i];
        if (cw != 0.f) {   // uniform branch across block: skips OOB/masked corners
            if (i + 16 < NLG * 4)
                __builtin_prefetch(&valb[(size_t)s_cl[i + 16] * DIMC + e], 0, 1);
            acc = fmaf(cw, (float)valb[(size_t)s_cl[i] * DIMC + e], acc);
        }
    }
    outpre[(size_t)q * DIMC + e] = (__bf16)acc;
}

// ---------------- host launcher ----------------
extern "C" void kernel_launch(void* const* d_in, const int* in_sizes, int n_in,
                              void* d_out, int out_size, void* d_ws, size_t ws_size,
                              hipStream_t stream) {
    const float* queries = (const float*)d_in[0];
    const float* img     = (const float*)d_in[1];   // [NC][256][HW] channel-major
    const float* ref     = (const float*)d_in[2];   // [NC][NQ][2]
    const unsigned char* vmask = (const unsigned char*)d_in[3];
    const float* Woff = (const float*)d_in[4];
    const float* boff = (const float*)d_in[5];
    const float* Ww   = (const float*)d_in[6];
    const float* bw   = (const float*)d_in[7];
    const float* Wv   = (const float*)d_in[8];
    const float* bv   = (const float*)d_in[9];
    const float* Wo   = (const float*)d_in[10];
    const float* bo   = (const float*)d_in[11];
    (void)in_sizes; (void)n_in; (void)out_size; (void)ws_size;

    char* ws = (char*)d_ws;
    size_t cur = 0;
    auto take = [&](size_t bytes) {
        void* p = ws + cur;
        cur = (cur + bytes + 255) & ~(size_t)255;
        return p;
    };
    __bf16* imgT   = (__bf16*)take((size_t)NCAM * HWC * DIMC * 2);  // [cam][hw][ch]
    __bf16* valb   = (__bf16*)take((size_t)NCAM * HWC * DIMC * 2);  // [cam][hw][e]
    __bf16* qb     = (__bf16*)take((size_t)NQC * DIMC * 2);
    __bf16* Wvt    = (__bf16*)take((size_t)DIMC * DIMC * 2);        // [n][k]
    __bf16* Wqt    = (__bf16*)take((size_t)NQO * DIMC * 2);         // [n][k]
    float*  biasq  = (float*) take((size_t)NQO * 4);
    float*  qout   = (float*) take((size_t)NQC * NQO * 4);
    __bf16* Wot    = (__bf16*)take((size_t)DIMC * DIMC * 2);        // [n][k]
    __bf16* outpre = (__bf16*)take((size_t)NQC * DIMC * 2);

    {   // conversions / packing / transposes
        int n = NQC * DIMC;
        cvt_f32_bf16<<<(n + 255) / 256, 256, 0, stream>>>(queries, qb, n);
        n = DIMC * DIMC;
        cvt_transpose_w<<<(n + 255) / 256, 256, 0, stream>>>(Wv, Wvt, DIMC, DIMC);
        cvt_transpose_w<<<(n + 255) / 256, 256, 0, stream>>>(Wo, Wot, DIMC, DIMC);
        n = NQO * DIMC;
        pack_qweights_t<<<(n + 255) / 256, 256, 0, stream>>>(Woff, Ww, boff, bw, Wqt, biasq);
        dim3 tg(HWC / 32, DIMC / 32, NCAM);
        transpose_cvt_img<<<tg, 256, 0, stream>>>(img, imgT);
    }
    {   // values[c][hw][e] = imgT[c][hw][:] @ Wv + bv   (dominant GEMM, bf16 out)
        dim3 grid(HWC / 128, DIMC / 64, NCAM);
        gemm_nt_wmma<true><<<grid, 256, 0, stream>>>(
            imgT, Wvt, bv, (void*)valb, HWC, DIMC,
            (long long)HWC * DIMC, (long long)HWC * DIMC);
    }
    {   // fused offsets+logits projection: qout[q][576]
        dim3 grid((NQC + 127) / 128, NQO / 64, 1);
        gemm_nt_wmma<false><<<grid, 256, 0, stream>>>(
            qb, Wqt, biasq, (void*)qout, NQC, NQO, 0, 0);
    }
    // softmax + deformable bilinear gather + head/point accumulation
    sample_accum<<<NQC, 256, 0, stream>>>(qout, ref, vmask, valb, outpre);
    {   // output projection -> fp32 d_out
        dim3 grid((NQC + 127) / 128, DIMC / 64, 1);
        gemm_nt_wmma<false><<<grid, 256, 0, stream>>>(
            outpre, Wot, bo, d_out, NQC, DIMC, 0, 0);
    }
}
